// AttentivePooling_28389733826838
// MI455X (gfx1250) — compile-verified, matrix operand order checked by reading, not run on hardware
//
#include <hip/hip_runtime.h>
#include <hip/hip_bf16.h>
#include <math.h>

typedef __attribute__((ext_vector_type(16))) __bf16 v16bf;
typedef __attribute__((ext_vector_type(8)))  float  v8f;

static constexpr int HDIM  = 128;   // feature dim
static constexpr int HHALF = 64;    // hidden dim
// LDS staging of W1 in B-fragment layout:
// 16 tiles (4 k-chunks x 4 n-tiles), each tile = 32 lanes x 16 bf16 = 512 elems
static constexpr int W1_FRAG_ELEMS = 16 * 512;   // 8192 bf16 = 16 KB

// ---------------------------------------------------------------------------
// Kernel 1: s[i] = tanh(x[i] @ W1 + b1) @ W2 + b2   via v_wmma_f32_16x16x32_bf16
// One wave per 16-row tile, 8 waves (256 threads) per block.
// ---------------------------------------------------------------------------
__global__ __launch_bounds__(256)
void attn_score_wmma_kernel(const float* __restrict__ x,
                            const float* __restrict__ W1,
                            const float* __restrict__ b1,
                            const float* __restrict__ W2,
                            const float* __restrict__ b2,
                            float* __restrict__ s,
                            int n_nodes)
{
    __shared__ __bf16 lds_w1[W1_FRAG_ELEMS];

    const int tid = threadIdx.x;

    // Stage W1 (f32 row-major [128][64]) into LDS, pre-swizzled into the
    // ISA 16-bit B-matrix (32x16) fragment layout so each lane later reads
    // its full fragment as one contiguous 32B ds load.
    for (int idx = tid; idx < W1_FRAG_ELEMS; idx += 256) {
        const int tile  = idx >> 9;        // 0..15
        const int rem   = idx & 511;
        const int flane = rem >> 4;        // fragment lane 0..31
        const int e     = rem & 15;        // element within fragment
        const int kc    = tile >> 2;       // k-chunk 0..3 (K step 32)
        const int nt    = tile & 3;        // n-tile  0..3 (N step 16)
        const int khalf = flane >> 4;
        const int col   = nt * 16 + (flane & 15);
        const int g     = e >> 3;
        const int r     = e & 7;
        const int K     = kc * 32 + g * 16 + khalf * 8 + r;
        lds_w1[idx] = (__bf16)W1[K * HHALF + col];
    }
    __syncthreads();

    const int wave = tid >> 5;
    const int lane = tid & 31;
    const int m0   = (blockIdx.x * 8 + wave) * 16;   // tile row base
    if (m0 >= n_nodes) return;                       // wave-uniform (N % 16 == 0)

    const int mloc  = lane & 15;
    const int khalf = lane >> 4;
    const float* __restrict__ xrow = x + (size_t)(m0 + mloc) * HDIM;

    v8f c0 = {}, c1 = {}, c2 = {}, c3 = {};

#pragma unroll
    for (int kc = 0; kc < 4; ++kc) {
        // A-fragment: 16x32 bf16 tile of x. Per ISA layout this lane holds
        // K = kc*32 + khalf*8 + {0..7} and + {16..23}, row = m0 + mloc.
        const int kbase = kc * 32 + khalf * 8;
        const float4 a0 = *(const float4*)(xrow + kbase +  0);
        const float4 a1 = *(const float4*)(xrow + kbase +  4);
        const float4 a2 = *(const float4*)(xrow + kbase + 16);
        const float4 a3 = *(const float4*)(xrow + kbase + 20);
        v16bf a;
        a[0]  = (__bf16)a0.x; a[1]  = (__bf16)a0.y; a[2]  = (__bf16)a0.z; a[3]  = (__bf16)a0.w;
        a[4]  = (__bf16)a1.x; a[5]  = (__bf16)a1.y; a[6]  = (__bf16)a1.z; a[7]  = (__bf16)a1.w;
        a[8]  = (__bf16)a2.x; a[9]  = (__bf16)a2.y; a[10] = (__bf16)a2.z; a[11] = (__bf16)a2.w;
        a[12] = (__bf16)a3.x; a[13] = (__bf16)a3.y; a[14] = (__bf16)a3.z; a[15] = (__bf16)a3.w;

        const __bf16* tb = &lds_w1[(kc * 4) * 512 + lane * 16];
        const v16bf b0 = *(const v16bf*)(tb + 0 * 512);
        const v16bf b1f = *(const v16bf*)(tb + 1 * 512);
        const v16bf b2f = *(const v16bf*)(tb + 2 * 512);
        const v16bf b3f = *(const v16bf*)(tb + 3 * 512);

        c0 = __builtin_amdgcn_wmma_f32_16x16x32_bf16(false, a, false, b0,  (short)0, c0, false, false);
        c1 = __builtin_amdgcn_wmma_f32_16x16x32_bf16(false, a, false, b1f, (short)0, c1, false, false);
        c2 = __builtin_amdgcn_wmma_f32_16x16x32_bf16(false, a, false, b2f, (short)0, c2, false, false);
        c3 = __builtin_amdgcn_wmma_f32_16x16x32_bf16(false, a, false, b3f, (short)0, c3, false, false);
    }

    // Epilogue. C/D layout: VGPR v -> row M = v + 8*khalf, column N = mloc
    // within each 16-wide n-tile. Each lane holds one column element of 8 rows
    // per n-tile; reduce tanh(c+b1)*W2 across the 16 lanes of each half-wave.
    const float bias2 = b2[0];
    float partial[8];
#pragma unroll
    for (int v = 0; v < 8; ++v) {
        float acc = 0.0f;
#pragma unroll
        for (int nt = 0; nt < 4; ++nt) {
            const int col = nt * 16 + mloc;
            const float cv = (nt == 0) ? c0[v] : (nt == 1) ? c1[v] : (nt == 2) ? c2[v] : c3[v];
            const float h  = tanhf(cv + b1[col]);
            acc += h * W2[col];
        }
#pragma unroll
        for (int off = 8; off > 0; off >>= 1)
            acc += __shfl_xor(acc, off, 16);
        partial[v] = acc;
    }
    if (mloc == 0) {
        const int rbase = m0 + khalf * 8;
#pragma unroll
        for (int v = 0; v < 8; ++v)
            s[rbase + v] = partial[v] + bias2;
    }
}

// ---------------------------------------------------------------------------
// Kernel 2: segment offsets via binary search on the sorted batch array.
// offsets[g] = first i with batch[i] >= g ; offsets[G] = N. Deterministic.
// ---------------------------------------------------------------------------
__global__ __launch_bounds__(256)
void seg_offsets_kernel(const int* __restrict__ batch,
                        int* __restrict__ offsets,
                        int n_nodes, int n_graphs)
{
    const int g = blockIdx.x * blockDim.x + threadIdx.x;
    if (g > n_graphs) return;
    if (g == n_graphs) { offsets[g] = n_nodes; return; }
    int lo = 0, hi = n_nodes;
    while (lo < hi) {
        const int mid = (lo + hi) >> 1;
        if (batch[mid] < g) lo = mid + 1; else hi = mid;
    }
    offsets[g] = lo;
}

// ---------------------------------------------------------------------------
// Kernel 3: per-graph softmax + weighted pooling. One wave per graph,
// 8 waves per block. Atomic-free -> bitwise deterministic.
// ---------------------------------------------------------------------------
__global__ __launch_bounds__(256)
void seg_softmax_pool_kernel(const float* __restrict__ x,
                             const float* __restrict__ s,
                             const int* __restrict__ offsets,
                             float* __restrict__ pooled,   // [G,128]
                             float* __restrict__ attn,     // [N]
                             int n_graphs)
{
    const int wave = threadIdx.x >> 5;
    const int lane = threadIdx.x & 31;
    const int g = blockIdx.x * 8 + wave;
    if (g >= n_graphs) return;

    const int beg = offsets[g];
    const int end = offsets[g + 1];

    // Phase 1: segment max (lane-strided over nodes)
    float m = -INFINITY;
    for (int i = beg + lane; i < end; i += 32) m = fmaxf(m, s[i]);
#pragma unroll
    for (int off = 16; off > 0; off >>= 1) m = fmaxf(m, __shfl_xor(m, off, 32));

    // Phase 2: segment sum of exp
    float d = 0.0f;
    for (int i = beg + lane; i < end; i += 32) d += __expf(s[i] - m);
#pragma unroll
    for (int off = 16; off > 0; off >>= 1) d += __shfl_xor(d, off, 32);
    const float inv_d = (end > beg) ? (1.0f / d) : 0.0f;

    // Phase 3: weights + pooled sum; all 32 lanes stream each 128-f32 row.
    const int dim0 = lane * 4;
    float4 acc = make_float4(0.f, 0.f, 0.f, 0.f);
    for (int i = beg; i < end; ++i) {
        if (i + 8 < end)
            __builtin_prefetch(x + (size_t)(i + 8) * HDIM + dim0, 0, 0);
        const float w = __expf(s[i] - m) * inv_d;
        if (lane == 0) attn[i] = w;
        const float4 xv = *(const float4*)(x + (size_t)i * HDIM + dim0);
        acc.x += w * xv.x; acc.y += w * xv.y;
        acc.z += w * xv.z; acc.w += w * xv.w;
    }
    *(float4*)(pooled + (size_t)g * HDIM + dim0) = acc;
}

// ---------------------------------------------------------------------------
// Launch
// inputs: x[N*128] f32, batch[N] i32, W1[128*64] f32, b1[64] f32,
//         W2[64] f32, b2[1] f32, num_graphs (scalar, unread on host)
// output: concat( x_pooled[G*128], attn_weights[N] ) f32
// ---------------------------------------------------------------------------
extern "C" void kernel_launch(void* const* d_in, const int* in_sizes, int n_in,
                              void* d_out, int out_size, void* d_ws, size_t ws_size,
                              hipStream_t stream)
{
    const float* x     = (const float*)d_in[0];
    const int*   batch = (const int*)  d_in[1];
    const float* W1    = (const float*)d_in[2];
    const float* b1    = (const float*)d_in[3];
    const float* W2    = (const float*)d_in[4];
    const float* b2    = (const float*)d_in[5];

    const int N = in_sizes[0] / HDIM;                 // 1,000,000
    const int G = (int)(((long)out_size - N) / HDIM); // 32,768 (from out layout)

    float* pooled = (float*)d_out;                    // [G*128]
    float* attn   = pooled + (size_t)G * HDIM;        // [N]

    // workspace: s[N] f32, offsets[G+1] i32  (~4.13 MB)
    float* s       = (float*)d_ws;
    int*   offsets = (int*)((char*)d_ws + (size_t)N * sizeof(float));

    const int score_blocks = (N + 127) / 128;         // 8 tiles of 16 rows / block
    attn_score_wmma_kernel<<<score_blocks, 256, 0, stream>>>(x, W1, b1, W2, b2, s, N);

    const int off_blocks = (G + 1 + 255) / 256;
    seg_offsets_kernel<<<off_blocks, 256, 0, stream>>>(batch, offsets, N, G);

    const int pool_blocks = (G + 7) / 8;              // one wave per graph
    seg_softmax_pool_kernel<<<pool_blocks, 256, 0, stream>>>(x, s, offsets, pooled, attn, G);
}
